// CAM_6949257085456
// MI455X (gfx1250) — matrix operand order repaired and checked by reading, not compiled, not run on hardware
//
#include <hip/hip_runtime.h>

// ---- problem dims ----
#define BB    512
#define TT    16
#define DAin  512
#define DVRAW 25088
#define MROWS (BB*TT)          // 8192

typedef __attribute__((ext_vector_type(16))) __bf16 v16bf;
typedef __attribute__((ext_vector_type(8)))  float  v8f;
typedef __attribute__((__vector_size__(4 * sizeof(int)))) int v4i;

// ---- optional gfx1250 async global->LDS path (compile-safe guard) ----
#if defined(__AMDGCN__) && __has_builtin(__builtin_amdgcn_global_load_async_to_lds_b128) && __has_builtin(__builtin_amdgcn_s_wait_asynccnt)
#define ASYNC_LDS 1
// builtin signature (from hipcc diagnostic): (global v4i*, shared v4i*, imm offset, imm cpol)
#define AS_GLOBAL(p) ((__attribute__((address_space(1))) v4i*)((__attribute__((address_space(1))) void*)(p)))
#define AS_LOCAL(p)  ((__attribute__((address_space(3))) v4i*)((__attribute__((address_space(3))) void*)(p)))
#else
#define ASYNC_LDS 0
#endif

// ---- bf16 fragment builders (native v_cvt_pk_bf16_f32 via (__bf16) casts) ----
// A-frag (16-bit A 16x32): lo lanes hold K = k0+[0..7],[16..23]; hi lanes +8.
__device__ __forceinline__ v16bf load_frag_a(const float* __restrict__ row, int k0, int half) {
  const float4* p0 = (const float4*)(row + k0 + half * 8);
  const float4* p1 = (const float4*)(row + k0 + 16 + half * 8);
  float4 x0 = p0[0], x1 = p0[1], x2 = p1[0], x3 = p1[1];
  v16bf r;
  r[0]  = (__bf16)x0.x; r[1]  = (__bf16)x0.y; r[2]  = (__bf16)x0.z; r[3]  = (__bf16)x0.w;
  r[4]  = (__bf16)x1.x; r[5]  = (__bf16)x1.y; r[6]  = (__bf16)x1.z; r[7]  = (__bf16)x1.w;
  r[8]  = (__bf16)x2.x; r[9]  = (__bf16)x2.y; r[10] = (__bf16)x2.z; r[11] = (__bf16)x2.w;
  r[12] = (__bf16)x3.x; r[13] = (__bf16)x3.y; r[14] = (__bf16)x3.z; r[15] = (__bf16)x3.w;
  return r;
}
// B-frag (16-bit B 32x16): lo lanes hold K = k0+[0..7],[16..23]; hi lanes +8; N = lane&15.
__device__ __forceinline__ v16bf load_frag_b(const float* __restrict__ row, int k0, int half) {
  const float4* p0 = (const float4*)(row + k0 + half * 8);
  const float4* p1 = (const float4*)(row + k0 + 16 + half * 8);
  float4 x0 = p0[0], x1 = p0[1], x2 = p1[0], x3 = p1[1];
  v16bf r;
  r[0]  = (__bf16)x0.x; r[1]  = (__bf16)x0.y; r[2]  = (__bf16)x0.z; r[3]  = (__bf16)x0.w;
  r[4]  = (__bf16)x1.x; r[5]  = (__bf16)x1.y; r[6]  = (__bf16)x1.z; r[7]  = (__bf16)x1.w;
  r[8]  = (__bf16)x2.x; r[9]  = (__bf16)x2.y; r[10] = (__bf16)x2.z; r[11] = (__bf16)x2.w;
  r[12] = (__bf16)x3.x; r[13] = (__bf16)x3.y; r[14] = (__bf16)x3.z; r[15] = (__bf16)x3.w;
  return r;
}

// C[row, colOff+n] = sum_k A[row,k]*W[n,k] + bias[n]
// Block = 256 threads = 8 waves. Block covers a 32-row M tile x (128*JW) cols.
// Each wave: 2 A-frags (M sub-tiles) x JW B-frags (nTile = wave + 8*j) -> 2*JW WMMAs/K-step.
// K must be a multiple of 32; N must equal 128*JW; M a multiple of 32.
template <int JW>
__global__ __launch_bounds__(256) void gemm_bf16_wmma(
    const float* __restrict__ A, int lda,
    const float* __restrict__ W, int ldw,
    const float* __restrict__ bias,
    float* __restrict__ C, int ldc, int colOff,
    int K)
{
  const int lane = threadIdx.x & 31;
  const int wave = threadIdx.x >> 5;
  const int half = lane >> 4;
  const int l16  = lane & 15;

  const float* Arow0 = A + ((size_t)blockIdx.x * 32 + l16) * (size_t)lda;
  const float* Arow1 = Arow0 + (size_t)16 * lda;

  const float* Wrow[JW];
  int nCol[JW];
#pragma unroll
  for (int j = 0; j < JW; ++j) {
    nCol[j] = (wave + 8 * j) * 16 + l16;
    Wrow[j] = W + (size_t)nCol[j] * (size_t)ldw;
  }

  v8f acc[2][JW];
#pragma unroll
  for (int mi = 0; mi < 2; ++mi)
#pragma unroll
    for (int j = 0; j < JW; ++j)
      acc[mi][j] = (v8f){0.f, 0.f, 0.f, 0.f, 0.f, 0.f, 0.f, 0.f};

  for (int k0 = 0; k0 < K; k0 += 32) {
    v16bf a0 = load_frag_a(Arow0, k0, half);
    v16bf a1 = load_frag_a(Arow1, k0, half);
#pragma unroll
    for (int j = 0; j < JW; ++j) {
      v16bf b = load_frag_b(Wrow[j], k0, half);
      acc[0][j] = __builtin_amdgcn_wmma_f32_16x16x32_bf16(false, a0, false, b,
                                                          (short)0, acc[0][j], false, false);
      acc[1][j] = __builtin_amdgcn_wmma_f32_16x16x32_bf16(false, a1, false, b,
                                                          (short)0, acc[1][j], false, false);
    }
  }

  // C layout: VGPR v, lanes 0-15: M=v, lanes 16-31: M=v+8; N = l16 within tile.
#pragma unroll
  for (int mi = 0; mi < 2; ++mi) {
    const size_t baseRow = (size_t)blockIdx.x * 32 + mi * 16 + half * 8;
#pragma unroll
    for (int j = 0; j < JW; ++j) {
      const float bn = bias[nCol[j]];
#pragma unroll
      for (int v = 0; v < 8; ++v) {
        C[(baseRow + v) * (size_t)ldc + colOff + nCol[j]] = acc[mi][j][v] + bn;
      }
    }
  }
}

// Fused CAM attention branch, one block per batch element, fp32.
// thread = (branch<<7) | c ; branch 0 = audio (cols 0..127), 1 = vis (cols 128..255).
__global__ __launch_bounds__(256) void branch_kernel(
    const float* __restrict__ avf,     // [8192, 256]
    const float* __restrict__ W_affa,  // [16,16]
    const float* __restrict__ W_affv,  // [16,16]
    const float* __restrict__ W_a,     // [32,16]
    const float* __restrict__ W_ca,    // [32,256]
    const float* __restrict__ W_ha,    // [16,32]
    const float* __restrict__ W_hv,    // [16,32]
    float* __restrict__ feats)         // [8192, 256]
{
  __shared__ float s_avf[16][256];   // 16 KB
  __shared__ float s_Wca[32][256];   // 32 KB
  __shared__ float s_aff[2][16][16];
  __shared__ float s_Wa[32][16];
  __shared__ float s_Wh[2][16][32];

  const int b   = blockIdx.x;
  const int tid = threadIdx.x;

#if ASYNC_LDS
  // Stage avf tile (16 KB) and W_ca (32 KB) via async global->LDS (ASYNCcnt path).
  {
    const float* gsrc = avf + (size_t)b * 16 * 256;
    for (int i = tid * 4; i < 16 * 256; i += 256 * 4)
      __builtin_amdgcn_global_load_async_to_lds_b128(AS_GLOBAL(gsrc + i),
                                                     AS_LOCAL(&s_avf[0][0] + i), 0, 0);
    for (int i = tid * 4; i < 32 * 256; i += 256 * 4)
      __builtin_amdgcn_global_load_async_to_lds_b128(AS_GLOBAL(W_ca + i),
                                                     AS_LOCAL(&s_Wca[0][0] + i), 0, 0);
  }
#else
  for (int i = tid; i < 16 * 256; i += 256) (&s_avf[0][0])[i] = avf[(size_t)b * 16 * 256 + i];
  for (int i = tid; i < 32 * 256; i += 256) (&s_Wca[0][0])[i] = W_ca[i];
#endif
  { // 256 elems each, one per thread
    (&s_aff[0][0][0])[tid] = W_affa[tid];
    (&s_aff[1][0][0])[tid] = W_affv[tid];
  }
  for (int i = tid; i < 32 * 16; i += 256) {
    (&s_Wa[0][0])[i]    = W_a[i];
    (&s_Wh[0][0][0])[i] = W_ha[i];
    (&s_Wh[1][0][0])[i] = W_hv[i];
  }
#if ASYNC_LDS
  __builtin_amdgcn_s_wait_asynccnt(0);  // each wave drains its own async ops pre-barrier
#endif
  __syncthreads();

  const int branch = tid >> 7;
  const int c      = tid & 127;
  const int fcol   = branch * 128 + c;

  // a_t[c][o] = sum_t fts[t][c] * W_aff[o][t]
  float at[16];
#pragma unroll
  for (int o = 0; o < 16; ++o) {
    float s = 0.f;
#pragma unroll
    for (int t = 0; t < 16; ++t) s += s_avf[t][fcol] * s_aff[branch][o][t];
    at[o] = s;
  }

  // H[c][k] = sum_j tanh(SCALE * sum_t avf[t][j]*at[t]) * W_ca[k][j]
  float H[32];
#pragma unroll
  for (int k = 0; k < 32; ++k) H[k] = 0.f;
  for (int j = 0; j < 256; ++j) {
    float s = 0.f;
#pragma unroll
    for (int t = 0; t < 16; ++t) s += s_avf[t][j] * at[t];
    float attv = tanhf(s * 0.0625f);   // SCALE = 1/sqrt(2*128) = 1/16
#pragma unroll
    for (int k = 0; k < 32; ++k) H[k] += attv * s_Wca[k][j];
  }
  // + W_a(fts.T), relu
#pragma unroll
  for (int k = 0; k < 32; ++k) {
    float s = 0.f;
#pragma unroll
    for (int t = 0; t < 16; ++t) s += s_avf[t][fcol] * s_Wa[k][t];
    H[k] = fmaxf(H[k] + s, 0.f);
  }
  // out[t][c] = sum_k H[k]*W_h[t][k] + fts[t][c]
#pragma unroll
  for (int t = 0; t < 16; ++t) {
    float s = 0.f;
#pragma unroll
    for (int k = 0; k < 32; ++k) s += H[k] * s_Wh[branch][t][k];
    feats[((size_t)b * 16 + t) * 256 + fcol] = s + s_avf[t][fcol];
  }
}

// Final 128 -> 1 heads for both regressors.
__global__ __launch_bounds__(256) void head_kernel(
    const float* __restrict__ hv, const float* __restrict__ ha,
    const float* __restrict__ Wv2, const float* __restrict__ bv2,
    const float* __restrict__ Wa2, const float* __restrict__ ba2,
    float* __restrict__ out)
{
  const int i = blockIdx.x * blockDim.x + threadIdx.x;  // 0..8191
  float sv = 0.f, sa = 0.f;
#pragma unroll 4
  for (int k = 0; k < 128; ++k) {
    sv += hv[(size_t)i * 128 + k] * Wv2[k];
    sa += ha[(size_t)i * 128 + k] * Wa2[k];
  }
  out[i]         = sv + bv2[0];
  out[MROWS + i] = sa + ba2[0];
}

extern "C" void kernel_launch(void* const* d_in, const int* in_sizes, int n_in,
                              void* d_out, int out_size, void* d_ws, size_t ws_size,
                              hipStream_t stream) {
  const float* f1     = (const float*)d_in[0];
  const float* f2     = (const float*)d_in[1];
  const float* W_enc1 = (const float*)d_in[2];
  const float* b_enc1 = (const float*)d_in[3];
  const float* W_enc2 = (const float*)d_in[4];
  const float* b_enc2 = (const float*)d_in[5];
  const float* W_red  = (const float*)d_in[6];
  const float* b_red  = (const float*)d_in[7];
  const float* W_affa = (const float*)d_in[8];
  const float* W_affv = (const float*)d_in[9];
  const float* W_a    = (const float*)d_in[10];
  const float* W_ca   = (const float*)d_in[11];
  const float* W_ha   = (const float*)d_in[12];
  const float* W_hv   = (const float*)d_in[13];
  const float* Wv1    = (const float*)d_in[14];
  const float* bv1    = (const float*)d_in[15];
  const float* Wv2    = (const float*)d_in[16];
  const float* bv2    = (const float*)d_in[17];
  const float* Wa1    = (const float*)d_in[18];
  const float* ba1    = (const float*)d_in[19];
  const float* Wa2    = (const float*)d_in[20];
  const float* ba2    = (const float*)d_in[21];

  // workspace layout (fp32): 8192*(512+256+256+128+128) = 10.5M floats = 40 MB
  float* ws      = (float*)d_ws;
  float* vis_mid = ws;                               // [8192, 512]
  float* avf     = vis_mid + (size_t)MROWS * 512;    // [8192, 256] (aud | vis)
  float* feats   = avf     + (size_t)MROWS * 256;    // [8192, 256]
  float* h_v     = feats   + (size_t)MROWS * 256;    // [8192, 128]
  float* h_a     = h_v     + (size_t)MROWS * 128;    // [8192, 128]

  // 1) vis_mid = f2 @ W_red^T + b_red   (dominant GEMM, K=25088, N=512)
  gemm_bf16_wmma<4><<<dim3(MROWS / 32), 256, 0, stream>>>(
      f2, DVRAW, W_red, DVRAW, b_red, vis_mid, 512, 0, DVRAW);
  // 2) avf[:, :128] = f1 @ W_enc1^T + b_enc1   (K=512, N=128)
  gemm_bf16_wmma<1><<<dim3(MROWS / 32), 256, 0, stream>>>(
      f1, DAin, W_enc1, DAin, b_enc1, avf, 256, 0, DAin);
  // 3) avf[:, 128:] = vis_mid @ W_enc2^T + b_enc2  (K=512, N=128)
  gemm_bf16_wmma<1><<<dim3(MROWS / 32), 256, 0, stream>>>(
      vis_mid, 512, W_enc2, 512, b_enc2, avf, 256, 128, 512);
  // 4) fused attention branches -> feats
  branch_kernel<<<dim3(BB), 256, 0, stream>>>(
      avf, W_affa, W_affv, W_a, W_ca, W_ha, W_hv, feats);
  // 5/6) regressor hidden layers (K=256, N=128)
  gemm_bf16_wmma<1><<<dim3(MROWS / 32), 256, 0, stream>>>(
      feats, 256, Wv1, 256, bv1, h_v, 128, 0, 256);
  gemm_bf16_wmma<1><<<dim3(MROWS / 32), 256, 0, stream>>>(
      feats, 256, Wa1, 256, ba1, h_a, 128, 0, 256);
  // 7) heads -> d_out = [vouts (8192) | aouts (8192)]
  head_kernel<<<dim3(MROWS / 256), 256, 0, stream>>>(
      h_v, h_a, Wv2, bv2, Wa2, ba2, (float*)d_out);
}
